// CausalSelfAttention_5111011082610
// MI455X (gfx1250) — matrix operand order repaired and checked by reading, not compiled, not run on hardware
//
#include <hip/hip_runtime.h>
#include <type_traits>

// ---------------------------------------------------------------------------
// CDNA5 (gfx1250) causal self-attention.
//   - LDS-tiled, register-pipelined GEMMs: 128x128x64 block tile, 8 waves,
//     each wave 32x64 output (2 A-frags x 4 B-frags reused -> 16 wmma / K-step)
//   - V projection stores V transposed [B,H,D,T] so flash-attn P@V B-fragments
//     are contiguous global_load_b128
//   - flash attention: one wave per 16 query rows; 64-key blocks (softmax
//     bookkeeping amortized 2x); mask compares only on the single diagonal
//     block; row-sums of P via WMMA against a ones-matrix
// ---------------------------------------------------------------------------

typedef __attribute__((ext_vector_type(16))) _Float16 v16h;
typedef __attribute__((ext_vector_type(8)))  _Float16 v8h;
typedef __attribute__((ext_vector_type(8)))  float    v8f;
typedef __attribute__((ext_vector_type(4)))  float    v4f;

#define NB 2          // batch
#define TS 2048       // sequence length
#define CE 1024       // embed dim
#define NH 16         // heads
#define DH 64         // head dim
#define MROWS (NB*TS) // 4096 GEMM rows

static __device__ __forceinline__ v8f wmma_f16(v16h a, v16h b, v8f c) {
  // D = A(16x32 f16) * B(32x16 f16) + C(16x16 f32)
  return __builtin_amdgcn_wmma_f32_16x16x32_f16(false, a, false, b,
                                                (short)0, c, false, false);
}

// ---------------------------------------------------------------------------
// Tiled GEMM:  Out = A @ W + bias
//   A_F32    : A is f32 [MROWS, CE] (x), else f16 [MROWS, CE] (attention y)
//   OUT_MODE : 0 -> f32 [MROWS, CE]          (output projection)
//              1 -> f16 scattered [B,H,T,D]  (Q, K)
//              2 -> f16 scattered [B,H,D,T]  (V transposed for P@V fragments)
// ---------------------------------------------------------------------------
template <bool A_F32, int OUT_MODE>
__global__ __launch_bounds__(256) void gemm_tiled(
    const void* __restrict__ Ap, const float* __restrict__ W,
    const float* __restrict__ bias, void* __restrict__ Op) {
  __shared__ alignas(16) _Float16 Ash[128][72];  // [m][k], padded stride
  __shared__ alignas(16) _Float16 Bsh[128][72];  // [n][k] (transposed W panel)

  const int tid  = threadIdx.x;
  const int wv   = tid >> 5;
  const int lane = tid & 31;
  const int half = lane >> 4;
  const int lmod = lane & 15;

  const int mBase = (int)blockIdx.y * 128;
  const int nBase = (int)blockIdx.x * 128;
  const int mPair  = wv & 3;   // 2 M-subtiles each
  const int nGroup = wv >> 2;  // 4 N-subtiles each

  const float*    Af = (const float*)Ap;
  const _Float16* Ah = (const _Float16*)Ap;

  v4f aF[8];   // staging (A_F32):  8 x float4 = 32 floats / thread
  v8h aH[4];   // staging (!A_F32): 4 x v8h    = 32 halfs  / thread
  v4f bF[8];   // staging W:        8 x float4 = 32 floats / thread

  auto load_stage = [&](int k0) {
    if constexpr (A_F32) {
#pragma unroll
      for (int i = 0; i < 8; ++i) {
        const int li = tid + i * 256;                 // 0..2047 v4f chunks
        const int r = li >> 4, cc = (li & 15) << 2;   // 128 rows x 16 chunks
        aF[i] = *(const v4f*)(Af + (size_t)(mBase + r) * CE + k0 + cc);
      }
    } else {
#pragma unroll
      for (int i = 0; i < 4; ++i) {
        const int li = tid + i * 256;                 // 0..1023 v8h chunks
        const int r = li >> 3, cc = (li & 7) << 3;    // 128 rows x 8 chunks
        aH[i] = *(const v8h*)(Ah + (size_t)(mBase + r) * CE + k0 + cc);
      }
    }
#pragma unroll
    for (int i = 0; i < 8; ++i) {
      const int li = tid + i * 256;                   // 0..2047 v4f chunks
      const int r = li >> 5, cc = (li & 31) << 2;     // 64 k-rows x 32 chunks
      bF[i] = *(const v4f*)(W + (size_t)(k0 + r) * CE + nBase + cc);
    }
  };

  auto stage_to_lds = [&]() {
    if constexpr (A_F32) {
#pragma unroll
      for (int i = 0; i < 8; ++i) {
        const int li = tid + i * 256;
        const int r = li >> 4, cc = (li & 15) << 2;
        _Float16* d = &Ash[r][cc];
        d[0] = (_Float16)aF[i][0]; d[1] = (_Float16)aF[i][1];
        d[2] = (_Float16)aF[i][2]; d[3] = (_Float16)aF[i][3];
      }
    } else {
#pragma unroll
      for (int i = 0; i < 4; ++i) {
        const int li = tid + i * 256;
        const int r = li >> 3, cc = (li & 7) << 3;
        *(v8h*)&Ash[r][cc] = aH[i];
      }
    }
#pragma unroll
    for (int i = 0; i < 8; ++i) {
      const int li = tid + i * 256;
      const int r = li >> 5, cc = (li & 31) << 2;     // k=r, n=cc..cc+3
#pragma unroll
      for (int u = 0; u < 4; ++u)
        Bsh[cc + u][r] = (_Float16)bF[i][u];          // transpose into LDS
    }
  };

  v8f acc[2][4] = {};

  load_stage(0);
  for (int k0 = 0; k0 < CE; k0 += 64) {
    __syncthreads();                 // previous tile fully consumed
    stage_to_lds();
    __syncthreads();
    if (k0 + 64 < CE) load_stage(k0 + 64);   // overlap with compute
    if (k0 + 128 < CE) {                     // L2 prefetch two tiles ahead
      if constexpr (A_F32)
        __builtin_prefetch(Af + (size_t)(mBase + (tid & 127)) * CE + k0 + 128, 0, 0);
      else
        __builtin_prefetch(Ah + (size_t)(mBase + (tid & 127)) * CE + k0 + 128, 0, 0);
      __builtin_prefetch(W + (size_t)(k0 + 128 + (tid >> 2)) * CE + nBase, 0, 0);
    }
#pragma unroll
    for (int kk = 0; kk < 64; kk += 32) {
      v16h a[2];
#pragma unroll
      for (int ms = 0; ms < 2; ++ms) {
        const _Float16* p0 = &Ash[(mPair * 2 + ms) * 16 + lmod][kk + half * 8];
        v8h c0 = *(const v8h*)p0;
        v8h c1 = *(const v8h*)(p0 + 16);
#pragma unroll
        for (int i = 0; i < 8; ++i) { a[ms][i] = c0[i]; a[ms][8 + i] = c1[i]; }
      }
#pragma unroll
      for (int ns = 0; ns < 4; ++ns) {
        const int n = (nGroup * 4 + ns) * 16 + lmod;
        const _Float16* p0 = &Bsh[n][kk + half * 8];
        v8h c0 = *(const v8h*)p0;
        v8h c1 = *(const v8h*)(p0 + 16);
        v16h b;
#pragma unroll
        for (int i = 0; i < 8; ++i) { b[i] = c0[i]; b[8 + i] = c1[i]; }
#pragma unroll
        for (int ms = 0; ms < 2; ++ms)
          acc[ms][ns] = wmma_f16(a[ms], b, acc[ms][ns]);
      }
    }
  }

  // epilogue
#pragma unroll
  for (int ns = 0; ns < 4; ++ns) {
    const int ncol = nBase + (nGroup * 4 + ns) * 16 + lmod;
    const float bn = bias[ncol];
#pragma unroll
    for (int ms = 0; ms < 2; ++ms) {
#pragma unroll
      for (int v = 0; v < 8; ++v) {
        const int m = mBase + (mPair * 2 + ms) * 16 + v + 8 * half;
        const float val = acc[ms][ns][v] + bn;
        if constexpr (OUT_MODE == 0) {
          ((float*)Op)[(size_t)m * CE + ncol] = val;
        } else {
          const int hh = ncol >> 6, dd = ncol & 63;
          const int bb = m >> 11, tt = m & (TS - 1);
          if constexpr (OUT_MODE == 1)   // [B,H,T,D]
            ((_Float16*)Op)[(((size_t)bb * NH + hh) * TS + tt) * DH + dd] =
                (_Float16)val;
          else                           // [B,H,D,T] (transposed V)
            ((_Float16*)Op)[(((size_t)bb * NH + hh) * DH + dd) * TS + tt] =
                (_Float16)val;
        }
      }
    }
  }
}

// ---------------------------------------------------------------------------
// Flash attention: one wave (block of 32) per (b,h, 16-row query tile).
// 64-key blocks: 8 wmmas for S, 2 for P row-sums (vs ones), 8 for P@V.
// Unmasked blocks take a compare-free fast path; the single diagonal block
// applies the causal mask.
// ---------------------------------------------------------------------------
__global__ __launch_bounds__(32) void attn_kernel(
    const _Float16* __restrict__ Q, const _Float16* __restrict__ K,
    const _Float16* __restrict__ Vt,   // [B,H,D,T]
    _Float16* __restrict__ Yh) {
  __shared__ alignas(16) _Float16 Plds[16 * 64];

  const int lane = threadIdx.x & 31;
  const int half = lane >> 4;
  const int lmod = lane & 15;
  const int nqb  = TS / 16;
  const int qb   = blockIdx.x % nqb;
  const int bh   = blockIdx.x / nqb;
  const int qBase = qb * 16;
  const _Float16* Qp  = Q  + (size_t)bh * TS * DH;
  const _Float16* Kp  = K  + (size_t)bh * TS * DH;
  const _Float16* Vtp = Vt + (size_t)bh * DH * TS;

  v16h qa[2];
  {
    const int qrow = qBase + lmod;
#pragma unroll
    for (int f = 0; f < 2; ++f) {
      const _Float16* p = Qp + (size_t)qrow * DH + f * 32 + half * 8;
      v8h c0 = *(const v8h*)(p);
      v8h c1 = *(const v8h*)(p + 16);
#pragma unroll
      for (int i = 0; i < 8; ++i) { qa[f][i] = c0[i]; qa[f][8 + i] = c1[i]; }
    }
  }

  v16h ones;
#pragma unroll
  for (int i = 0; i < 16; ++i) ones[i] = (_Float16)1.0f;

  v8f acc[4] = {};
  float mrow[8], lrow[8], alpha[8];
  const float NEG_INF = -__builtin_inff();
#pragma unroll
  for (int v = 0; v < 8; ++v) { mrow[v] = NEG_INF; lrow[v] = 0.f; }
  // exp2 domain: softmax(s/sqrt(d)) == 2^(s*cs - m2) normalized
  const float cs = 0.125f * 1.44269504089f;   // log2(e)/sqrt(64)

  auto process_block = [&](int j, auto masked_tag) {
    constexpr bool MASKED = decltype(masked_tag)::value;
    // ---- S = Q K^T for keys [j, j+64) : 4 key sub-tiles x 2 d-halves ----
    v8f s[4];
#pragma unroll
    for (int t = 0; t < 4; ++t) {
      const int key = j + t * 16 + lmod;
      v8f sz = {};
#pragma unroll
      for (int f = 0; f < 2; ++f) {
        const _Float16* p = Kp + (size_t)key * DH + f * 32 + half * 8;
        v8h c0 = *(const v8h*)(p);
        v8h c1 = *(const v8h*)(p + 16);
        v16h kb;
#pragma unroll
        for (int i = 0; i < 8; ++i) { kb[i] = c0[i]; kb[8 + i] = c1[i]; }
        sz = wmma_f16(qa[f], kb, sz);
      }
      s[t] = sz;
    }
    // ---- online softmax (row M = v + 8*half across a 16-lane group) ----
#pragma unroll
    for (int v = 0; v < 8; ++v) {
      const int M = v + 8 * half;
      float sv[4];
#pragma unroll
      for (int t = 0; t < 4; ++t) {
        if constexpr (MASKED) {
          const int query = qBase + M;
          sv[t] = (j + t * 16 + lmod <= query) ? (float)s[t][v] * cs : NEG_INF;
        } else {
          sv[t] = (float)s[t][v] * cs;
        }
      }
      float mx = fmaxf(fmaxf(sv[0], sv[1]), fmaxf(sv[2], sv[3]));
#pragma unroll
      for (int off = 1; off < 16; off <<= 1)
        mx = fmaxf(mx, __shfl_xor(mx, off, 16));
      const float mnew = fmaxf(mrow[v], mx);
      alpha[v] = exp2f(mrow[v] - mnew);
      mrow[v]  = mnew;
#pragma unroll
      for (int t = 0; t < 4; ++t)
        Plds[M * 64 + t * 16 + lmod] = (_Float16)exp2f(sv[t] - mnew);
#pragma unroll
      for (int t = 0; t < 4; ++t) acc[t][v] *= alpha[v];
    }
    __syncthreads();
    // ---- reload P as two A-fragments (k = keys 0..31 / 32..63) ----
    v16h pf[2];
#pragma unroll
    for (int f = 0; f < 2; ++f) {
      const _Float16* pp = &Plds[lmod * 64 + f * 32 + half * 8];
      v8h c0 = *(const v8h*)(pp);
      v8h c1 = *(const v8h*)(pp + 16);
#pragma unroll
      for (int i = 0; i < 8; ++i) { pf[f][i] = c0[i]; pf[f][8 + i] = c1[i]; }
    }
    __syncthreads();
    // ---- row sums of P via WMMA vs ones: D[M][*] = sum_k P[M][k] ----
    {
      v8f zero = {};
      v8f psum = wmma_f16(pf[1], ones, wmma_f16(pf[0], ones, zero));
#pragma unroll
      for (int v = 0; v < 8; ++v) lrow[v] = lrow[v] * alpha[v] + psum[v];
    }
    // ---- acc += P @ V (B-fragments contiguous from transposed V) ----
#pragma unroll
    for (int t = 0; t < 4; ++t) {
#pragma unroll
      for (int f = 0; f < 2; ++f) {
        const _Float16* p =
            Vtp + (size_t)(t * 16 + lmod) * TS + j + f * 32 + half * 8;
        v8h c0 = *(const v8h*)(p);
        v8h c1 = *(const v8h*)(p + 16);
        v16h vb;
#pragma unroll
        for (int i = 0; i < 8; ++i) { vb[i] = c0[i]; vb[8 + i] = c1[i]; }
        acc[t] = wmma_f16(pf[f], vb, acc[t]);
      }
    }
  };

  const int jDiag = ((qBase + 15) >> 6) << 6;   // single masked 64-key block
  for (int j = 0; j < jDiag; j += 64)
    process_block(j, std::false_type{});        // compare-free fast path
  process_block(jDiag, std::true_type{});       // diagonal block with mask

  const int b = bh / NH, h = bh % NH;
#pragma unroll
  for (int v = 0; v < 8; ++v) {
    const int M = v + 8 * half;
    const int trow = qBase + M;
    const float inv = 1.f / lrow[v];
#pragma unroll
    for (int t = 0; t < 4; ++t) {
      const int cc = h * DH + t * 16 + lmod;
      Yh[((size_t)b * TS + trow) * CE + cc] = (_Float16)((float)acc[t][v] * inv);
    }
  }
}

// ---------------------------------------------------------------------------
extern "C" void kernel_launch(void* const* d_in, const int* in_sizes, int n_in,
                              void* d_out, int out_size, void* d_ws, size_t ws_size,
                              hipStream_t stream) {
  const float* x  = (const float*)d_in[0];
  const float* Wq = (const float*)d_in[1];
  const float* bq = (const float*)d_in[2];
  const float* Wk = (const float*)d_in[3];
  const float* bk = (const float*)d_in[4];
  const float* Wv = (const float*)d_in[5];
  const float* bv = (const float*)d_in[6];
  const float* Wp = (const float*)d_in[7];
  const float* bp = (const float*)d_in[8];
  float* out = (float*)d_out;

  const size_t qkvElems = (size_t)NB * NH * TS * DH;   // 4,194,304 halfs each
  _Float16* Qh  = (_Float16*)d_ws;
  _Float16* Kh  = Qh + qkvElems;
  _Float16* Vth = Kh + qkvElems;                       // [B,H,D,T]
  _Float16* Yh  = Vth + qkvElems;                      // [B*T, C] f16

  dim3 ggrid(CE / 128, MROWS / 128);                   // (8, 32) = 256 blocks
  gemm_tiled<true,  1><<<ggrid, 256, 0, stream>>>(x,  Wq, bq, Qh);
  gemm_tiled<true,  1><<<ggrid, 256, 0, stream>>>(x,  Wk, bk, Kh);
  gemm_tiled<true,  2><<<ggrid, 256, 0, stream>>>(x,  Wv, bv, Vth);

  attn_kernel<<<NB * NH * (TS / 16), 32, 0, stream>>>(Qh, Kh, Vth, Yh);

  gemm_tiled<false, 0><<<ggrid, 256, 0, stream>>>(Yh, Wp, bp, out);
}
